// MetabolicBrain_39539468927736
// MI455X (gfx1250) — compile-verified, matrix operand order checked by reading, not run on hardware
//
#include <hip/hip_runtime.h>
#include <math.h>

// Problem dims
#define BB 12
#define TT 2048
#define DD 260
#define VV 256
#define RR 280
#define D3 780   // 3*D
#define D2 520   // 2*D

// tile counts
#define KT_D   9    // ceil(260/32), also covers K=280
#define NT_D3  49   // ceil(780/16)
#define NT_D   17   // ceil(260/16)
#define NT_R   18   // ceil(280/16)
#define NT_V   16   // 256/16

typedef __attribute__((ext_vector_type(16))) __bf16 v16bf;
typedef __attribute__((ext_vector_type(8)))  float  v8f;
typedef __attribute__((ext_vector_type(4))) unsigned int u32x4;
typedef __attribute__((ext_vector_type(4))) int i32x4;
typedef __attribute__((ext_vector_type(8))) int i32x8;

union ABFrag { v16bf v; uint4 q[2]; };

__device__ __forceinline__ unsigned short f2bf(float x) {
  unsigned int b = __float_as_uint(x);
  b += 0x7FFFu + ((b >> 16) & 1u);          // round-to-nearest-even
  return (unsigned short)(b >> 16);
}
__device__ __forceinline__ float bf2f(unsigned short h) {
  return __uint_as_float(((unsigned int)h) << 16);
}
__device__ __forceinline__ unsigned int pk2(float a, float b) {
  return (unsigned int)f2bf(a) | ((unsigned int)f2bf(b) << 16);
}
__device__ __forceinline__ float sigmoidf(float x) { return 1.0f / (1.0f + expf(-x)); }

__device__ __forceinline__ float wave_sum(float v) {
#pragma unroll
  for (int o = 16; o > 0; o >>= 1) v += __shfl_xor(v, o, 32);
  return v;
}
__device__ __forceinline__ float wave_max(float v) {
#pragma unroll
  for (int o = 16; o > 0; o >>= 1) v = fmaxf(v, __shfl_xor(v, o, 32));
  return v;
}

// ---- Tensor Data Mover: async DMA of a contiguous f32 block (global -> LDS) ----
// 1-row 2D descriptor per CDNA5 ISA ch.8 (count=1, type=image(2), data_size=4B).
__device__ __forceinline__ void tdm_load_f32(unsigned int ldsOff, const void* gsrc,
                                             unsigned int nElems) {
  unsigned long long ga = (unsigned long long)gsrc;
  u32x4 g0;
  g0[0] = 1u;                                            // count=1 (valid user D#)
  g0[1] = ldsOff;                                        // lds_addr (bytes)
  g0[2] = (unsigned int)(ga & 0xFFFFFFFFull);            // global_addr[31:0]
  g0[3] = (unsigned int)((ga >> 32) & 0x01FFFFFFull) | 0x80000000u; // addr[56:32]|type=2
  i32x8 g1;
  g1[0] = 0x00020000;                                    // workgroup_mask=0, data_size=2(4B)
  g1[1] = (int)((nElems & 0xFFFFu) << 16);               // tensor_dim0[15:0] -> bits 63:48
  g1[2] = (int)(((nElems >> 16) & 0xFFFFu) | (1u << 16));// tensor_dim0[31:16], tensor_dim1=1
  g1[3] = (int)((nElems & 0xFFFFu) << 16);               // tile_dim0 (bits 127:112)
  g1[4] = 1;                                             // tile_dim1=1, tile_dim2=0
  g1[5] = (int)nElems;                                   // tensor_dim0_stride[31:0]
  g1[6] = 0;
  g1[7] = 0;
  i32x4 z4 = {0, 0, 0, 0};
#if defined(__clang_major__) && (__clang_major__ >= 23)
  i32x8 z8 = {0, 0, 0, 0, 0, 0, 0, 0};
  __builtin_amdgcn_tensor_load_to_lds(g0, g1, z4, z4, z8, 0);
#else
  __builtin_amdgcn_tensor_load_to_lds(g0, g1, z4, z4, 0);
#endif
}

// Pack 8 consecutive K-values of one row into a uint4 of bf16 (zero-pad past K).
__device__ __forceinline__ uint4 pack8(const float* __restrict__ rowp, int kbase, int K) {
  uint4 q;
  if (kbase + 8 <= K) {
    const float4* p = (const float4*)(rowp + kbase);   // 16B-aligned by construction
    float4 a = p[0], b = p[1];
    q.x = pk2(a.x, a.y); q.y = pk2(a.z, a.w);
    q.z = pk2(b.x, b.y); q.w = pk2(b.z, b.w);
  } else {
    float v[8];
#pragma unroll
    for (int j = 0; j < 8; ++j) { int k = kbase + j; v[j] = (k < K) ? rowp[k] : 0.f; }
    q.x = pk2(v[0], v[1]); q.y = pk2(v[2], v[3]);
    q.z = pk2(v[4], v[5]); q.w = pk2(v[6], v[7]);
  }
  return q;
}

// Stage an f32 state matrix (rows x K, row stride ld) into WMMA A-fragment layout.
__device__ __forceinline__ void stage_afrag(uint4* dstQ, const float* __restrict__ src,
                                            int rows, int ld, int K, int Kt,
                                            int tid, int nth) {
  const int total = Kt * 64;
  for (int i = tid; i < total; i += nth) {
    int kt = i >> 6, r2 = i & 63;
    int lane2 = r2 >> 1, hg = r2 & 1;
    int row = lane2 & 15;
    int kbase = kt * 32 + hg * 16 + ((lane2 >> 4) << 3);
    uint4 q = {0u, 0u, 0u, 0u};
    if (row < rows) q = pack8(src + (long)row * ld, kbase, K);
    dstQ[i] = q;
  }
}

// Wave-level tile GEMM.  All KT B-fragments are preloaded into a register array
// so the loads issue as one clause and the wmma chain drains with staggered waits.
template <int KT>
__device__ __forceinline__ v8f wmma_tile_acc(const uint4* __restrict__ aQ,
                                             const uint4* __restrict__ bQ,
                                             int lane, v8f acc) {
  ABFrag fb[KT];
#pragma unroll
  for (int kt = 0; kt < KT; ++kt) {
    const int o = (kt * 32 + lane) * 2;
    fb[kt].q[0] = bQ[o];                                  // global_load_b128
    fb[kt].q[1] = bQ[o + 1];
  }
#pragma unroll
  for (int kt = 0; kt < KT; ++kt) {
    ABFrag fa;
    const int oa = (kt * 32 + lane) * 2;
    fa.q[0] = aQ[oa]; fa.q[1] = aQ[oa + 1];               // ds_load_b128
    acc = __builtin_amdgcn_wmma_f32_16x16x32_bf16(false, fa.v, false, fb[kt].v,
                                                  (short)0, acc, false, false);
  }
  return acc;
}

// ---------------- packing kernels (one-time, perf-irrelevant) ----------------
__global__ __launch_bounds__(256) void pack_bfrag(const float* __restrict__ src,
                                                  uint4* __restrict__ dst,
                                                  int N, int Ksrc, int koff, int Kpart,
                                                  int Nt, int Kt, int kmajor,
                                                  const float* __restrict__ invn) {
  const long total = (long)Nt * Kt * 64;
  for (long i = blockIdx.x * 256L + threadIdx.x; i < total; i += (long)gridDim.x * 256L) {
    int nt = (int)(i / (Kt * 64));
    int r  = (int)(i % (Kt * 64));
    int kt = r >> 6, r2 = r & 63;
    int lane2 = r2 >> 1, hg = r2 & 1;
    int n = nt * 16 + (lane2 & 15);
    int kbase = kt * 32 + hg * 16 + ((lane2 >> 4) << 3);
    float sc = 1.f;
    if (invn && n < N) sc = invn[n];
    float v[8];
#pragma unroll
    for (int j = 0; j < 8; ++j) {
      int k = kbase + j;
      float val = 0.f;
      if (n < N && k < Kpart) {
        val = kmajor ? src[(long)(k + koff) * N + n]
                     : src[(long)n * Ksrc + (k + koff)];
        val *= sc;
      }
      v[j] = val;
    }
    uint4 q;
    q.x = pk2(v[0], v[1]); q.y = pk2(v[2], v[3]);
    q.z = pk2(v[4], v[5]); q.w = pk2(v[6], v[7]);
    dst[i] = q;
  }
}
__global__ __launch_bounds__(256) void rownorm_inv(const float* __restrict__ src,
                                                   float* __restrict__ invn, int K) {
  __shared__ float red[256];
  const int n = blockIdx.x;
  float s = 0.f;
  for (int k = threadIdx.x; k < K; k += 256) { float v = src[(long)n * K + k]; s += v * v; }
  red[threadIdx.x] = s;
  __syncthreads();
  for (int o = 128; o > 0; o >>= 1) {
    if ((int)threadIdx.x < o) red[threadIdx.x] += red[threadIdx.x + o];
    __syncthreads();
  }
  if (threadIdx.x == 0) invn[n] = 1.0f / fmaxf(sqrtf(red[0]), 1e-12f);
}

// ---------------- kernel 1: xin = emb @ Wih^T + bih ----------------
__global__ __launch_bounds__(512) void xin_kernel(const int* __restrict__ x,
                                                  const float* __restrict__ soma_w,
                                                  const uint4* __restrict__ WihP,
                                                  const float* __restrict__ bih,
                                                  float* __restrict__ xin) {
  __shared__ __align__(16) uint4 afragQ[KT_D * 64];
  __shared__ int toks[16];
  const int m0 = blockIdx.x * 16;
  const int tid = threadIdx.x;
  if (tid < 16) toks[tid] = x[m0 + tid];
  __syncthreads();
  for (int i = tid; i < KT_D * 64; i += 512) {
    int kt = i >> 6, r2 = i & 63;
    int lane2 = r2 >> 1, hg = r2 & 1;
    int row = lane2 & 15;
    int kbase = kt * 32 + hg * 16 + ((lane2 >> 4) << 3);
    afragQ[i] = pack8(soma_w + (long)toks[row] * DD, kbase, DD);
  }
  __syncthreads();
  const int w = tid >> 5, lane = tid & 31;
  for (int nt = w; nt < NT_D3; nt += 16) {
    if (nt + 16 < NT_D3) __builtin_prefetch(WihP + (size_t)(nt + 16) * KT_D * 64, 0, 1);
    v8f acc = {};
    acc = wmma_tile_acc<KT_D>(afragQ, WihP + (size_t)nt * KT_D * 64, lane, acc);
    const int col = lane & 15, rb = (lane >> 4) << 3;
#pragma unroll
    for (int r2 = 0; r2 < 8; ++r2) {
      int row = rb + r2, n = nt * 16 + col;
      if (n < D3) xin[(long)(m0 + row) * D3 + n] = acc[r2] + bih[n];
    }
  }
}

// ---------------- kernel 2: sequential fast GRU scan (persistent, 1 WG) ----------------
__global__ __launch_bounds__(512) void fastgru_kernel(const float* __restrict__ h_f,
                                                      const uint4* __restrict__ WhhP,
                                                      const float* __restrict__ bhh,
                                                      const float* __restrict__ xin,
                                                      float* __restrict__ p_f,
                                                      float* __restrict__ out_hf) {
  __shared__ __align__(16) float h[BB * DD];
  __shared__ __align__(16) float hh[BB * D3];
  __shared__ __align__(16) uint4 afragQ[KT_D * 64];
  const int tid = threadIdx.x, w = tid >> 5, lane = tid & 31;
  for (int i = tid; i < BB * DD; i += 512) h[i] = h_f[i];
  __syncthreads();
  for (int t = 0; t < TT; ++t) {
    stage_afrag(afragQ, h, BB, DD, DD, KT_D, tid, 512);
    __syncthreads();
    for (int nt = w; nt < NT_D3; nt += 16) {
      if (nt + 16 < NT_D3) __builtin_prefetch(WhhP + (size_t)(nt + 16) * KT_D * 64, 0, 1);
      v8f acc = {};
      acc = wmma_tile_acc<KT_D>(afragQ, WhhP + (size_t)nt * KT_D * 64, lane, acc);
      const int col = lane & 15, rb = (lane >> 4) << 3;
#pragma unroll
      for (int r2 = 0; r2 < 8; ++r2) {
        int row = rb + r2, n = nt * 16 + col;
        if (row < BB && n < D3) hh[row * D3 + n] = acc[r2] + bhh[n];
      }
    }
    __syncthreads();
    for (int i = tid; i < BB * DD; i += 512) {
      int b = i / DD, j = i % DD;
      long base = ((long)b * TT + t) * D3;
      float xr = xin[base + j], xz = xin[base + DD + j], xn = xin[base + 2 * DD + j];
      float hr = hh[b * D3 + j], hz = hh[b * D3 + DD + j], hn = hh[b * D3 + 2 * DD + j];
      float r  = sigmoidf(xr + hr);
      float z  = sigmoidf(xz + hz);
      float nn = tanhf(xn + r * hn);
      float hv = (1.0f - z) * nn + z * h[i];
      h[i] = hv;
      p_f[((long)t * BB + b) * DD + j] = hv;
    }
    __syncthreads();
  }
  for (int i = tid; i < BB * DD; i += 512) out_hf[i] = h[i];
}

// ---------------- kernel 3: t_step scan (persistent, 1 WG, TDM-prefetched latent) ----
__global__ __launch_bounds__(512) void tstep_kernel(
    const float* __restrict__ h_mono_in, const int* __restrict__ surprise,
    const int* __restrict__ dyn, const float* __restrict__ p_f,
    const uint4* __restrict__ WihP, const uint4* __restrict__ WhhP,
    const float* __restrict__ bih, const float* __restrict__ bhh,
    const float* __restrict__ vgW, const float* __restrict__ vgb,
    const uint4* __restrict__ keysP, const uint4* __restrict__ valsP,
    const uint4* __restrict__ pfcA, const uint4* __restrict__ pfcB,
    const float* __restrict__ pfcb, const float* __restrict__ pfcg,
    const float* __restrict__ pfcbeta,
    float* __restrict__ outs, float* __restrict__ out_hm, float* __restrict__ out_sc) {
  __shared__ __align__(16) float hm[BB * DD];
  __shared__ __align__(16) float lat[BB * DD];
  __shared__ __align__(16) float scratch[6480];     // phase-overlaid
  __shared__ __align__(16) uint4 afragQ[KT_D * 64];
  __shared__ float vol[BB];
  __shared__ float rnorm[BB];
  __shared__ float acc2[2];                         // [0]=cost, [1]=will
  const int tid = threadIdx.x, w = tid >> 5, lane = tid & 31;
  const int ds = dyn[0];
  const float sb = logf(fmaxf(1.0f, (float)surprise[0]));
  const unsigned int latLds = (unsigned int)(unsigned long long)(const void*)lat;

  unsigned short* s1b  = (unsigned short*)scratch;           // r-gate preact (bf16)
  unsigned short* s2b  = s1b + BB * DD;                      // z-gate preact
  unsigned short* gnxb = s1b + 2 * BB * DD;                  // gi_n
  unsigned short* ghnb = s1b + 3 * BB * DD;                  // gh_n
  float* attn = scratch;                                     // [12][280] f32
  float* ctx  = scratch + 3360;                              // [12][260] f32
  float* zz   = scratch;                                     // [12][260] (attn dead)

  for (int i = tid; i < BB * DD; i += 512) hm[i] = h_mono_in[i];
  for (int i = tid; i < BB * DD; i += 512) lat[i] = p_f[i];   // latent for t=0
  if (tid == 0) { acc2[0] = 0.f; acc2[1] = 0.f; }
  __syncthreads();

  for (int t = 0; t < TT; ++t) {
    // fence the TDM that filled `lat` for this step (no-op at t==0)
    __builtin_amdgcn_s_wait_tensorcnt(0);
    __syncthreads();

    for (int s = 0; s < ds; ++s) {
      // volition gate vol = sigmoid(latent @ vg_W^T + vg_b + sb)
      if (w < BB) {
        float sum = 0.f;
        for (int k = lane; k < DD; k += 32) sum += lat[w * DD + k] * vgW[k];
        sum = wave_sum(sum);
        if (lane == 0) vol[w] = sigmoidf(sum + vgb[0] + sb);
      }
      __syncthreads();
      if (tid == 0) {
        float vm = 0.f;
        for (int b = 0; b < BB; ++b) vm += vol[b];
        vm *= (1.0f / BB);
        acc2[1] += vm;                  // will
        acc2[0] += vm * (0.12f / TT);   // cost
      }
      // grucell pass 1: gi = latent@Wih^T + bih
      stage_afrag(afragQ, lat, BB, DD, DD, KT_D, tid, 512);
      __syncthreads();
      for (int nt = w; nt < NT_D3; nt += 16) {
        if (nt + 16 < NT_D3) __builtin_prefetch(WihP + (size_t)(nt + 16) * KT_D * 64, 0, 1);
        v8f gi = {};
        gi = wmma_tile_acc<KT_D>(afragQ, WihP + (size_t)nt * KT_D * 64, lane, gi);
        const int col = lane & 15, rb = (lane >> 4) << 3;
#pragma unroll
        for (int r2 = 0; r2 < 8; ++r2) {
          int row = rb + r2, n = nt * 16 + col;
          if (row < BB && n < D3) {
            float a = gi[r2] + bih[n];
            if (n < DD)            s1b[row * DD + n]             = f2bf(a);
            else if (n < 2 * DD)   s2b[row * DD + (n - DD)]      = f2bf(a);
            else                   gnxb[row * DD + (n - 2 * DD)] = f2bf(a);
          }
        }
      }
      __syncthreads();
      // grucell pass 2: gh = hm@Whh^T + bhh, combine
      stage_afrag(afragQ, hm, BB, DD, DD, KT_D, tid, 512);
      __syncthreads();
      for (int nt = w; nt < NT_D3; nt += 16) {
        if (nt + 16 < NT_D3) __builtin_prefetch(WhhP + (size_t)(nt + 16) * KT_D * 64, 0, 1);
        v8f gh = {};
        gh = wmma_tile_acc<KT_D>(afragQ, WhhP + (size_t)nt * KT_D * 64, lane, gh);
        const int col = lane & 15, rb = (lane >> 4) << 3;
#pragma unroll
        for (int r2 = 0; r2 < 8; ++r2) {
          int row = rb + r2, n = nt * 16 + col;
          if (row < BB && n < D3) {
            float c = gh[r2] + bhh[n];
            if (n < DD)            { int o = row * DD + n;        s1b[o] = f2bf(bf2f(s1b[o]) + c); }
            else if (n < 2 * DD)   { int o = row * DD + (n - DD); s2b[o] = f2bf(bf2f(s2b[o]) + c); }
            else                   ghnb[row * DD + (n - 2 * DD)] = f2bf(c);
          }
        }
      }
      __syncthreads();
      // gate math + latent update
      for (int i = tid; i < BB * DD; i += 512) {
        int b = i / DD;
        float r  = sigmoidf(bf2f(s1b[i]));
        float z  = sigmoidf(bf2f(s2b[i]));
        float nn = tanhf(bf2f(gnxb[i]) + r * bf2f(ghnb[i]));
        float hv = (1.0f - z) * nn + z * hm[i];
        hm[i] = hv;
        lat[i] += hv * vol[b];
      }
      __syncthreads();
    }

    // attention scores = l2norm(latent) @ keys_n^T (row scale folded into scatter)
    if (w < BB) {
      float s = 0.f;
      for (int k = lane; k < DD; k += 32) { float v = lat[w * DD + k]; s += v * v; }
      s = wave_sum(s);
      if (lane == 0) rnorm[w] = 1.0f / fmaxf(sqrtf(s), 1e-12f);
    }
    __syncthreads();
    stage_afrag(afragQ, lat, BB, DD, DD, KT_D, tid, 512);
    __syncthreads();
    for (int nt = w; nt < NT_R; nt += 16) {
      v8f sc = {};
      sc = wmma_tile_acc<KT_D>(afragQ, keysP + (size_t)nt * KT_D * 64, lane, sc);
      const int col = lane & 15, rb = (lane >> 4) << 3;
#pragma unroll
      for (int r2 = 0; r2 < 8; ++r2) {
        int row = rb + r2, n = nt * 16 + col;
        if (row < BB && n < RR) attn[row * RR + n] = sc[r2] * rnorm[row];
      }
    }
    __syncthreads();
    // softmax per row (one wave per row)
    if (w < BB) {
      float ev[9];
      float mx = -3.0e38f;
      for (int k = lane; k < RR; k += 32) mx = fmaxf(mx, attn[w * RR + k]);
      mx = wave_max(mx);
      float sum = 0.f;
      int i2 = 0;
      for (int k = lane; k < RR; k += 32, ++i2) {
        float e = expf(attn[w * RR + k] - mx);
        ev[i2] = e; sum += e;
      }
      sum = wave_sum(sum);
      const float inv = 1.0f / sum;
      i2 = 0;
      for (int k = lane; k < RR; k += 32, ++i2) attn[w * RR + k] = ev[i2] * inv;
    }
    __syncthreads();
    // ctx = attn @ rel_vals
    stage_afrag(afragQ, attn, BB, RR, RR, KT_D, tid, 512);
    __syncthreads();
    for (int nt = w; nt < NT_D; nt += 16) {
      v8f cc = {};
      cc = wmma_tile_acc<KT_D>(afragQ, valsP + (size_t)nt * KT_D * 64, lane, cc);
      const int col = lane & 15, rb = (lane >> 4) << 3;
#pragma unroll
      for (int r2 = 0; r2 < 8; ++r2) {
        int row = rb + r2, n = nt * 16 + col;
        if (row < BB && n < DD) ctx[row * DD + n] = cc[r2];
      }
    }
    __syncthreads();
    // z = concat([latent, ctx]) @ pfc_W^T + pfc_b (two packed K halves)
    stage_afrag(afragQ, lat, BB, DD, DD, KT_D, tid, 512);
    __syncthreads();
    // `lat` is dead from here on: DMA next timestep's latent into it via TDM,
    // overlapping with the pfc GEMMs and LN/GELU below.
    if (w == 0 && t + 1 < TT)
      tdm_load_f32(latLds, p_f + (size_t)(t + 1) * BB * DD, BB * DD);
    v8f za0 = {}, za1 = {};
    {
      int cnt = 0;
      for (int nt = w; nt < NT_D; nt += 16) {
        const uint4* bQ = pfcA + (size_t)nt * KT_D * 64;
        if (cnt == 0) za0 = wmma_tile_acc<KT_D>(afragQ, bQ, lane, za0);
        else          za1 = wmma_tile_acc<KT_D>(afragQ, bQ, lane, za1);
        ++cnt;
      }
    }
    __syncthreads();
    stage_afrag(afragQ, ctx, BB, DD, DD, KT_D, tid, 512);
    __syncthreads();
    {
      int cnt = 0;
      for (int nt = w; nt < NT_D; nt += 16) {
        const uint4* bQ = pfcB + (size_t)nt * KT_D * 64;
        v8f zt = (cnt == 0) ? wmma_tile_acc<KT_D>(afragQ, bQ, lane, za0)
                            : wmma_tile_acc<KT_D>(afragQ, bQ, lane, za1);
        const int col = lane & 15, rb = (lane >> 4) << 3;
#pragma unroll
        for (int r2 = 0; r2 < 8; ++r2) {
          int row = rb + r2, n = nt * 16 + col;
          if (row < BB && n < DD) zz[row * DD + n] = zt[r2] + pfcb[n];
        }
        ++cnt;
      }
    }
    __syncthreads();
    // LayerNorm + exact GELU -> outs[B,T,D]
    if (w < BB) {
      float zv[9];
      float m = 0.f;
      int i2 = 0;
      for (int k = lane; k < DD; k += 32, ++i2) { zv[i2] = zz[w * DD + k]; m += zv[i2]; }
      m = wave_sum(m) * (1.0f / DD);
      float v = 0.f;
      i2 = 0;
      for (int k = lane; k < DD; k += 32, ++i2) { float d = zv[i2] - m; v += d * d; }
      v = wave_sum(v) * (1.0f / DD);
      const float rs = rsqrtf(v + 1e-5f);
      i2 = 0;
      for (int k = lane; k < DD; k += 32, ++i2) {
        float g = (zv[i2] - m) * rs * pfcg[k] + pfcbeta[k];
        float o = 0.5f * g * (1.0f + erff(g * 0.70710678f));
        outs[((long)w * TT + t) * DD + k] = o;
      }
    }
    __syncthreads();
  }

  for (int i = tid; i < BB * DD; i += 512) out_hm[i] = hm[i];
  if (tid == 0) {
    int dsm = (ds > 1) ? ds : 1;
    out_sc[0] = (float)ds;
    out_sc[1] = acc2[0];
    out_sc[2] = acc2[1] / ((float)TT * (float)dsm);
  }
}

// ---------------- kernel 4: logits = ln(outs) @ l2norm(soma_w)^T * gain ----------------
__global__ __launch_bounds__(256) void logits_kernel(const float* __restrict__ outs,
                                                     const float* __restrict__ on_g,
                                                     const float* __restrict__ on_b,
                                                     const uint4* __restrict__ nsomaP,
                                                     const float* __restrict__ gain,
                                                     float* __restrict__ logits) {
  __shared__ __align__(16) float a[16 * DD];
  __shared__ __align__(16) uint4 afragQ[KT_D * 64];
  __shared__ float mu[16], rs[16];
  const int m0 = blockIdx.x * 16;
  const int tid = threadIdx.x, w = tid >> 5, lane = tid & 31;
  for (int i = tid; i < 16 * DD; i += 256) {
    int r = i / DD, k = i % DD;
    a[i] = outs[(long)(m0 + r) * DD + k];
  }
  __syncthreads();
  if (tid < 16) {
    float m = 0.f;
    for (int k = 0; k < DD; ++k) m += a[tid * DD + k];
    m *= (1.0f / DD);
    float v = 0.f;
    for (int k = 0; k < DD; ++k) { float d = a[tid * DD + k] - m; v += d * d; }
    v *= (1.0f / DD);
    mu[tid] = m; rs[tid] = rsqrtf(v + 1e-5f);
  }
  __syncthreads();
  for (int i = tid; i < KT_D * 64; i += 256) {
    int kt = i >> 6, r2 = i & 63;
    int lane2 = r2 >> 1, hg = r2 & 1;
    int row = lane2 & 15;
    int kbase = kt * 32 + hg * 16 + ((lane2 >> 4) << 3);
    const float* rowp = a + row * DD;
    float m = mu[row], sc = rs[row];
    float v[8];
#pragma unroll
    for (int j = 0; j < 8; ++j) {
      int k = kbase + j;
      v[j] = (k < DD) ? ((rowp[k] - m) * sc * on_g[k] + on_b[k]) : 0.f;
    }
    uint4 q;
    q.x = pk2(v[0], v[1]); q.y = pk2(v[2], v[3]);
    q.z = pk2(v[4], v[5]); q.w = pk2(v[6], v[7]);
    afragQ[i] = q;
  }
  __syncthreads();
  const float gn = gain[0];
  for (int nt = w; nt < NT_V; nt += 8) {
    if (nt + 8 < NT_V) __builtin_prefetch(nsomaP + (size_t)(nt + 8) * KT_D * 64, 0, 1);
    v8f acc = {};
    acc = wmma_tile_acc<KT_D>(afragQ, nsomaP + (size_t)nt * KT_D * 64, lane, acc);
    const int col = lane & 15, rb = (lane >> 4) << 3;
#pragma unroll
    for (int r2 = 0; r2 < 8; ++r2) {
      int row = rb + r2, n = nt * 16 + col;
      logits[(long)(m0 + row) * VV + n] = acc[r2] * gn;
    }
  }
}

// ---------------- host launch ----------------
extern "C" void kernel_launch(void* const* d_in, const int* in_sizes, int n_in,
                              void* d_out, int out_size, void* d_ws, size_t ws_size,
                              hipStream_t stream) {
  (void)in_sizes; (void)n_in; (void)out_size; (void)ws_size;
  const int*   x        = (const int*)  d_in[0];
  const float* h_f      = (const float*)d_in[1];
  const float* h_mono   = (const float*)d_in[2];
  const int*   surprise = (const int*)  d_in[3];
  const int*   dynsteps = (const int*)  d_in[4];
  const float* soma_w   = (const float*)d_in[5];
  const float* gru_Wih  = (const float*)d_in[6];
  const float* gru_Whh  = (const float*)d_in[7];
  const float* gru_bih  = (const float*)d_in[8];
  const float* gru_bhh  = (const float*)d_in[9];
  const float* ws_Wih   = (const float*)d_in[10];
  const float* ws_Whh   = (const float*)d_in[11];
  const float* ws_bih   = (const float*)d_in[12];
  const float* ws_bhh   = (const float*)d_in[13];
  const float* vg_W     = (const float*)d_in[14];
  const float* vg_b     = (const float*)d_in[15];
  const float* rel_keys = (const float*)d_in[16];
  const float* rel_vals = (const float*)d_in[17];
  const float* pfc_W    = (const float*)d_in[18];
  const float* pfc_b    = (const float*)d_in[19];
  const float* pfc_g    = (const float*)d_in[20];
  const float* pfc_beta = (const float*)d_in[21];
  const float* on_g     = (const float*)d_in[22];
  const float* on_b     = (const float*)d_in[23];
  const float* gain     = (const float*)d_in[24];

  char* ws = (char*)d_ws;
  size_t off = 0;
  auto alloc = [&](size_t bytes) -> void* {
    void* p = ws + off;
    off += (bytes + 255) & ~(size_t)255;
    return p;
  };
  const size_t TILE = 512;  // halves per (nt,kt) tile
  float* xin  = (float*)alloc(sizeof(float) * (size_t)BB * TT * D3);
  float* p_f  = (float*)alloc(sizeof(float) * (size_t)TT * BB * DD);
  float* outs = (float*)alloc(sizeof(float) * (size_t)BB * TT * DD);
  uint4* WihP   = (uint4*)alloc(2 * (size_t)NT_D3 * KT_D * TILE);
  uint4* WhhP   = (uint4*)alloc(2 * (size_t)NT_D3 * KT_D * TILE);
  uint4* wsWihP = (uint4*)alloc(2 * (size_t)NT_D3 * KT_D * TILE);
  uint4* wsWhhP = (uint4*)alloc(2 * (size_t)NT_D3 * KT_D * TILE);
  uint4* keysP  = (uint4*)alloc(2 * (size_t)NT_R * KT_D * TILE);
  uint4* valsP  = (uint4*)alloc(2 * (size_t)NT_D * KT_D * TILE);
  uint4* pfcAP  = (uint4*)alloc(2 * (size_t)NT_D * KT_D * TILE);
  uint4* pfcBP  = (uint4*)alloc(2 * (size_t)NT_D * KT_D * TILE);
  uint4* nsomaP = (uint4*)alloc(2 * (size_t)NT_V * KT_D * TILE);
  float* invKeys = (float*)alloc(sizeof(float) * RR);
  float* invSoma = (float*)alloc(sizeof(float) * VV);

  float* out_logits = (float*)d_out;
  float* out_hf = out_logits + (size_t)BB * TT * VV;
  float* out_hm = out_hf + BB * DD;
  float* out_sc = out_hm + BB * DD;

  rownorm_inv<<<RR, 256, 0, stream>>>(rel_keys, invKeys, DD);
  rownorm_inv<<<VV, 256, 0, stream>>>(soma_w, invSoma, DD);

  pack_bfrag<<<256, 256, 0, stream>>>(gru_Wih, WihP,   D3, DD, 0, DD, NT_D3, KT_D, 0, nullptr);
  pack_bfrag<<<256, 256, 0, stream>>>(gru_Whh, WhhP,   D3, DD, 0, DD, NT_D3, KT_D, 0, nullptr);
  pack_bfrag<<<256, 256, 0, stream>>>(ws_Wih,  wsWihP, D3, DD, 0, DD, NT_D3, KT_D, 0, nullptr);
  pack_bfrag<<<256, 256, 0, stream>>>(ws_Whh,  wsWhhP, D3, DD, 0, DD, NT_D3, KT_D, 0, nullptr);
  pack_bfrag<<<256, 256, 0, stream>>>(rel_keys, keysP, RR, DD, 0, DD, NT_R, KT_D, 0, invKeys);
  pack_bfrag<<<256, 256, 0, stream>>>(rel_vals, valsP, DD, RR, 0, RR, NT_D, KT_D, 1, nullptr);
  pack_bfrag<<<256, 256, 0, stream>>>(pfc_W, pfcAP, DD, D2, 0,  DD, NT_D, KT_D, 0, nullptr);
  pack_bfrag<<<256, 256, 0, stream>>>(pfc_W, pfcBP, DD, D2, DD, DD, NT_D, KT_D, 0, nullptr);
  pack_bfrag<<<256, 256, 0, stream>>>(soma_w, nsomaP, VV, DD, 0, DD, NT_V, KT_D, 0, invSoma);

  xin_kernel<<<(BB * TT) / 16, 512, 0, stream>>>(x, soma_w, WihP, gru_bih, xin);
  fastgru_kernel<<<1, 512, 0, stream>>>(h_f, WhhP, gru_bhh, xin, p_f, out_hf);
  tstep_kernel<<<1, 512, 0, stream>>>(h_mono, surprise, dynsteps, p_f,
                                      wsWihP, wsWhhP, ws_bih, ws_bhh, vg_W, vg_b,
                                      keysP, valsP, pfcAP, pfcBP,
                                      pfc_b, pfc_g, pfc_beta,
                                      outs, out_hm, out_sc);
  logits_kernel<<<(BB * TT) / 16, 256, 0, stream>>>(outs, on_g, on_b, nsomaP, gain,
                                                    out_logits);
}
